// CNCAgg_14078902796571
// MI455X (gfx1250) — matrix operand order repaired and checked by reading, not compiled, not run on hardware
//
#include <hip/hip_runtime.h>

#define BB   32
#define NN   4096
#define CC   256
#define OUTC 512
#define WW   64
#define EPSF 1e-5f

typedef __attribute__((ext_vector_type(16))) __bf16 v16bf;
typedef __attribute__((ext_vector_type(8)))  __bf16 v8bf;
typedef __attribute__((ext_vector_type(8)))  float  v8f;

// Native f32 -> bf16 (RNE via fptrunc; backend picks the hardware cvt op).
__device__ __forceinline__ __bf16 f2bf(float f) { return (__bf16)f; }

__device__ __forceinline__ v16bf cvt16(v8f x, v8f y) {
  v8bf xb = __builtin_convertvector(x, v8bf);
  v8bf yb = __builtin_convertvector(y, v8bf);
  v16bf a;
#pragma unroll
  for (int i = 0; i < 8; ++i) { a[i] = xb[i]; a[i + 8] = yb[i]; }
  return a;
}

// Load a 16x32 bf16 A-fragment row slice from f32 memory.
// Lane layout (contiguous K): row = lane%16, lanes<16 hold K {0..15},
// lanes>=16 hold K {16..31}. kb8 = (lane<16)?0:8. Two contiguous 32B loads.
__device__ __forceinline__ v16bf load_a_f32(const float* __restrict__ row, int kb8) {
  v8f x = *(const v8f*)(row + 2 * kb8);
  v8f y = *(const v8f*)(row + 2 * kb8 + 8);
  return cvt16(x, y);
}

__device__ __forceinline__ v8f wmma_bf16(v16bf a, v16bf b, v8f c) {
  return __builtin_amdgcn_wmma_f32_16x16x32_bf16(false, a, false, b, (short)0, c,
                                                 false, false);
}

// ---------------------------------------------------------------------------
// Kernel 1: WeightNet (3 -> 64 -> 64 -> 64), layers 2&3 via WMMA.
// Each wave owns 32 consecutive points; stages activations in LDS in the
// B-fragment order (col = point, K = channel). Output is written to global
// memory directly in WMMA-B-fragment layout for the big GEMM:
//   wsB[((b*128 + kt)*4 + wt)*32*16 + lane*16 + i]
//   value = weights[b][w = wt*16 + (lane&15)][n = kt*32 + (lane<16?0:16) + i]
// ---------------------------------------------------------------------------
__global__ __launch_bounds__(128) void weightnet_kernel(
    const float* __restrict__ xyz,
    const float* __restrict__ w1, const float* __restrict__ b1,
    const float* __restrict__ g1, const float* __restrict__ be1,
    const float* __restrict__ m1, const float* __restrict__ v1,
    const float* __restrict__ w2, const float* __restrict__ b2,
    const float* __restrict__ g2, const float* __restrict__ be2,
    const float* __restrict__ m2, const float* __restrict__ v2,
    const float* __restrict__ w3, const float* __restrict__ b3,
    const float* __restrict__ g3, const float* __restrict__ be3,
    const float* __restrict__ m3, const float* __restrict__ v3,
    unsigned short* __restrict__ wsB_u16)
{
  __bf16* __restrict__ wsB = (__bf16*)wsB_u16;
  __shared__ __bf16 hbuf[2][4][32][WW];  // [buf][wave][point][channel]
  __shared__ float  sbn[6][WW];          // s1,sh1,s2,sh2,s3,sh3 (BN+bias folded)

  const int tid  = threadIdx.x;
  const int wv   = tid >> 5;
  const int lane = tid & 31;

  if (tid < WW) {
    int w = tid;
    float s1 = g1[w] * rsqrtf(v1[w] + EPSF);
    sbn[0][w] = s1; sbn[1][w] = (b1[w] - m1[w]) * s1 + be1[w];
    float s2 = g2[w] * rsqrtf(v2[w] + EPSF);
    sbn[2][w] = s2; sbn[3][w] = (b2[w] - m2[w]) * s2 + be2[w];
    float s3 = g3[w] * rsqrtf(v3[w] + EPSF);
    sbn[4][w] = s3; sbn[5][w] = (b3[w] - m3[w]) * s3 + be3[w];
  }
  __syncthreads();

  // ---- layer 1 (3 -> 64): scalar per point, write bf16 to LDS -------------
  const int p = blockIdx.x * 128 + tid;            // global point in [0, B*N)
  const float x0 = xyz[p * 3 + 0], x1 = xyz[p * 3 + 1], x2 = xyz[p * 3 + 2];
  for (int w = 0; w < WW; ++w) {
    float acc = fmaf(w1[w * 3 + 0], x0,
                fmaf(w1[w * 3 + 1], x1, w1[w * 3 + 2] * x2));
    float y = fmaxf(fmaf(acc, sbn[0][w], sbn[1][w]), 0.0f);
    hbuf[0][wv][lane][w] = f2bf(y);
  }
  __syncthreads();

  const int lw   = lane & 15;
  const int kb8  = (lane < 16) ? 0 : 8;
  const int kb16 = (lane < 16) ? 0 : 16;
  const int rofs = (lane < 16) ? 0 : 8;

  // ---- layer 2 (64 -> 64) via WMMA: A = w2 tile, B = activations ----------
#pragma unroll
  for (int pt = 0; pt < 2; ++pt) {
#pragma unroll
    for (int mt = 0; mt < 4; ++mt) {
      v8f acc = {};
#pragma unroll
      for (int ks = 0; ks < 2; ++ks) {
        int k0 = ks * 32;
        v16bf a = load_a_f32(w2 + (mt * 16 + lw) * WW + k0, kb8);
        v16bf b = *(const v16bf*)&hbuf[0][wv][pt * 16 + lw][k0 + kb16];
        acc = wmma_bf16(a, b, acc);
      }
#pragma unroll
      for (int r = 0; r < 8; ++r) {
        int m = mt * 16 + rofs + r;
        float y = fmaxf(fmaf(acc[r], sbn[2][m], sbn[3][m]), 0.0f);
        hbuf[1][wv][pt * 16 + lw][m] = f2bf(y);
      }
    }
  }
  __syncthreads();

  // ---- layer 3 (64 -> 64) via WMMA, store global in B-fragment layout -----
  const int bidx  = (blockIdx.x * 128) / NN;
  const int nbase = (blockIdx.x * 128 + wv * 32) % NN;
  const int kt    = nbase >> 5;                    // k-tile (32 points each)
  __bf16* basep = wsB + (((size_t)bidx * 128 + kt) * 4) * 32 * 16;
#pragma unroll
  for (int pt = 0; pt < 2; ++pt) {
#pragma unroll
    for (int mt = 0; mt < 4; ++mt) {
      v8f acc = {};
#pragma unroll
      for (int ks = 0; ks < 2; ++ks) {
        int k0 = ks * 32;
        v16bf a = load_a_f32(w3 + (mt * 16 + lw) * WW + k0, kb8);
        v16bf b = *(const v16bf*)&hbuf[1][wv][pt * 16 + lw][k0 + kb16];
        acc = wmma_bf16(a, b, acc);
      }
      int kk    = pt * 16 + lw;                    // point within k-tile
      int laneq = (kk >= 16) ? 16 : 0;
      int ii    = kk & 15;
#pragma unroll
      for (int r = 0; r < 8; ++r) {
        int wch = mt * 16 + rofs + r;
        float y = fmaxf(fmaf(acc[r], sbn[4][wch], sbn[5][wch]), 0.0f);
        int wt = wch >> 4;
        int lp = (wch & 15) + laneq;
        basep[((size_t)wt * 32 + lp) * 16 + ii] = f2bf(y);
      }
    }
  }
}

// ---------------------------------------------------------------------------
// Kernel 2: agg[b] = (feature[b]/N) x weightsT[b]   (256x4096)x(4096x64)
// One wave per 16-row strip, all 4 col-tiles -> feature read exactly once.
// ---------------------------------------------------------------------------
__global__ __launch_bounds__(256) void agg_gemm_kernel(
    const float* __restrict__ feat, const unsigned short* __restrict__ wsB_u16,
    float* __restrict__ agg)
{
  const __bf16* __restrict__ wsB = (const __bf16*)wsB_u16;
  const int tid = threadIdx.x, wv = tid >> 5, lane = tid & 31;
  const int b  = blockIdx.x >> 1;
  const int ct = ((blockIdx.x & 1) << 3) + wv;     // 0..15 row tile
  const int lw = lane & 15;
  const int kb8  = (lane < 16) ? 0 : 8;
  const int rofs = (lane < 16) ? 0 : 8;
  const int c0 = ct << 4;

  const float*  arow  = feat + ((size_t)b * CC + (c0 + lw)) * NN;
  const __bf16* bbase = wsB + (size_t)b * (NN / 32) * 4 * 32 * 16;

  v8f acc0 = {}, acc1 = {}, acc2 = {}, acc3 = {};
#pragma unroll 2
  for (int kt = 0; kt < NN / 32; ++kt) {
    // prefetch the A stream ~8 k-tiles ahead (gfx1250 global_prefetch_b8)
    __builtin_prefetch(arow + kt * 32 + 256, 0, 0);
    v16bf a = load_a_f32(arow + kt * 32, kb8);
    const __bf16* bp = bbase + ((size_t)kt * 4 * 32 + lane) * 16;
    v16bf b0 = *(const v16bf*)(bp + 0 * 32 * 16);
    v16bf b1 = *(const v16bf*)(bp + 1 * 32 * 16);
    v16bf b2 = *(const v16bf*)(bp + 2 * 32 * 16);
    v16bf b3 = *(const v16bf*)(bp + 3 * 32 * 16);
    acc0 = wmma_bf16(a, b0, acc0);
    acc1 = wmma_bf16(a, b1, acc1);
    acc2 = wmma_bf16(a, b2, acc2);
    acc3 = wmma_bf16(a, b3, acc3);
  }
  const float sc = 1.0f / NN;                      // feat = feature / N folded here
  float* ab = agg + (size_t)b * (CC * WW);
#pragma unroll
  for (int r = 0; r < 8; ++r) {
    int c = c0 + rofs + r;
    ab[c * WW + lw +  0] = acc0[r] * sc;
    ab[c * WW + lw + 16] = acc1[r] * sc;
    ab[c * WW + lw + 32] = acc2[r] * sc;
    ab[c * WW + lw + 48] = acc3[r] * sc;
  }
}

// ---------------------------------------------------------------------------
// Kernel 3: out = relu(bn(wf @ agg + bf))  ==  (32x16384)x(16384x512) GEMM
// wf is row-major (OUT, 16384): K is contiguous -> B-fragment is one 64B load.
// ---------------------------------------------------------------------------
__global__ __launch_bounds__(128) void final_gemm_kernel(
    const float* __restrict__ agg, const float* __restrict__ wf,
    const float* __restrict__ bf,  const float* __restrict__ gf,
    const float* __restrict__ bef, const float* __restrict__ mf,
    const float* __restrict__ vf,  float* __restrict__ out)
{
  const int tid = threadIdx.x, wv = tid >> 5, lane = tid & 31;
  const int tile = blockIdx.x * 4 + wv;            // 0..63
  const int m0 = (tile & 1) << 4;
  const int o0 = (tile >> 1) << 4;
  const int lw = lane & 15;
  const int kb8  = (lane < 16) ? 0 : 8;
  const int kb16 = (lane < 16) ? 0 : 16;
  const int rofs = (lane < 16) ? 0 : 8;
  const int K = CC * WW;

  const float* arow = agg + (size_t)(m0 + lw) * K;
  const float* brow = wf  + (size_t)(o0 + lw) * K;

  v8f acc = {};
#pragma unroll 2
  for (int k0 = 0; k0 < K; k0 += 32) {
    // prefetch the streamed wf rows ahead (32 MB stream sets the roofline)
    __builtin_prefetch(brow + k0 + 256, 0, 0);
    v16bf a = load_a_f32(arow + k0, kb8);
    v8f x = *(const v8f*)(brow + k0 + kb16);
    v8f y = *(const v8f*)(brow + k0 + kb16 + 8);
    v16bf bfrag = cvt16(x, y);
    acc = wmma_bf16(a, bfrag, acc);
  }
  int o = o0 + lw;
  float s  = gf[o] * rsqrtf(vf[o] + EPSF);
  float sh = (bf[o] - mf[o]) * s + bef[o];
#pragma unroll
  for (int r = 0; r < 8; ++r) {
    int m = m0 + rofs + r;
    if (m < BB) out[(size_t)m * OUTC + o] = fmaxf(fmaf(acc[r], s, sh), 0.0f);
  }
}

extern "C" void kernel_launch(void* const* d_in, const int* in_sizes, int n_in,
                              void* d_out, int out_size, void* d_ws, size_t ws_size,
                              hipStream_t stream) {
  (void)in_sizes; (void)n_in; (void)out_size; (void)ws_size;
  const float* xyz     = (const float*)d_in[0];
  const float* feature = (const float*)d_in[1];
  // d_in[2] = idx : unused by the reference computation
  const float* w1 = (const float*)d_in[3];  const float* b1  = (const float*)d_in[4];
  const float* g1 = (const float*)d_in[5];  const float* be1 = (const float*)d_in[6];
  const float* m1 = (const float*)d_in[7];  const float* v1  = (const float*)d_in[8];
  const float* w2 = (const float*)d_in[9];  const float* b2  = (const float*)d_in[10];
  const float* g2 = (const float*)d_in[11]; const float* be2 = (const float*)d_in[12];
  const float* m2 = (const float*)d_in[13]; const float* v2  = (const float*)d_in[14];
  const float* w3 = (const float*)d_in[15]; const float* b3  = (const float*)d_in[16];
  const float* g3 = (const float*)d_in[17]; const float* be3 = (const float*)d_in[18];
  const float* m3 = (const float*)d_in[19]; const float* v3  = (const float*)d_in[20];
  const float* wf = (const float*)d_in[21]; const float* bfv = (const float*)d_in[22];
  const float* gf = (const float*)d_in[23]; const float* bef = (const float*)d_in[24];
  const float* mf = (const float*)d_in[25]; const float* vf  = (const float*)d_in[26];
  float* out = (float*)d_out;

  unsigned short* wsB = (unsigned short*)d_ws;                   // 16 MB bf16 frags
  float* agg = (float*)((char*)d_ws + (size_t)BB * NN * WW * 2); // 2 MB f32

  weightnet_kernel<<<dim3(BB * NN / 128), dim3(128), 0, stream>>>(
      xyz, w1, b1, g1, be1, m1, v1, w2, b2, g2, be2, m2, v2,
      w3, b3, g3, be3, m3, v3, wsB);
  agg_gemm_kernel<<<dim3(BB * 2), dim3(256), 0, stream>>>(feature, wsB, agg);
  final_gemm_kernel<<<dim3(16), dim3(128), 0, stream>>>(
      agg, wf, bfv, gf, bef, mf, vf, out);
}
